// TransformerBlock_36773509988428
// MI455X (gfx1250) — matrix operand order repaired and checked by reading, not compiled, not run on hardware
//
#include <hip/hip_runtime.h>

// Transformer block for MI455X (gfx1250), wave32 + WMMA f16.
// One wave == one batch item (S=16 rows == WMMA M-tile).
// Weights staged in LDS in WMMA B-fragment order; transposes via per-wave
// LDS scratch with disjoint, bank-conflict-aware regions (no VGPR spills).
// All WMMA paths run with EXEC all-ones (branchless zero-pad reads).

#define BATCH 16384
#define SEQ   16
#define DM    64
#define NH    4
#define DFF   256
#define WAVES 16
#define THREADS (WAVES * 32)
#define NWG   128

typedef __attribute__((ext_vector_type(16))) _Float16 v16h;
typedef __attribute__((ext_vector_type(8)))  _Float16 v8h;
typedef __attribute__((ext_vector_type(8)))  float    v8f;

#define NTILES   192          // 96 QKV + 32 Wp + 32 W1 + 32 W2, 1KB each (f16)
#define ROWP     72           // padded row stride (halves), 16x64 row-major stage [0,1152)
#define VT_OFF   1152         // V^T stage: 64 rows x 16 halves, stride VTP    [1152,2176)
#define VTP      16
#define ATTN_OFF 2176         // attn 16x16 stage, stride ATTNP                [2176,2560)
#define ATTNP    24
#define ZERO_OFF 2560         // 32 halves of zeros for branchless zero-pad reads
#define RELUP    40           // padded row stride for 16x32 relu stage (in rowStage area)
#define SCRH     2624         // per-wave scratch halves

static __device__ __forceinline__ void fence_lds() {
  // Same-wave LDS is processed in order; this pins compiler ordering and
  // drains DScnt between produce/consume phases.
  asm volatile("s_wait_dscnt 0" ::: "memory");
}

static __device__ __forceinline__ v16h join8(v8h lo, v8h hi) {
  return __builtin_shufflevector(lo, hi, 0,1,2,3,4,5,6,7,8,9,10,11,12,13,14,15);
}

static __device__ __forceinline__ v8h zero8() {
  v8h z;
#pragma unroll
  for (int i = 0; i < 8; ++i) z[i] = (_Float16)0.0f;
  return z;
}

// B fragment from a pre-swizzled LDS weight tile: lane's 16 halves contiguous.
static __device__ __forceinline__ v16h ld_bfrag(const _Float16* tile, int lane) {
  const v8h* p = (const v8h*)(tile + lane * 16);
  return join8(p[0], p[1]);
}

// A fragment (16x32 f16) from row-major 16x64 stage (row stride ROWP).
// A layout: lane holds row nl; e0..15 -> K = 32kt + 16*half + e.
static __device__ __forceinline__ v16h ld_afrag(const _Float16* rs, int nl, int half, int kt) {
  const _Float16* p = rs + nl * ROWP + kt * 32 + half * 16;
  return join8(*(const v8h*)p, *(const v8h*)(p + 8));
}

// B fragment of K^T from row-major K stage: lane n = nl (token t),
// K index = 32kt + 16*half + e  -> 16 contiguous halves of row nl.
static __device__ __forceinline__ v16h ld_ktb(const _Float16* rs, int nl, int half, int kt) {
  const _Float16* p = rs + nl * ROWP + kt * 32 + half * 16;
  return join8(*(const v8h*)p, *(const v8h*)(p + 8));
}

static __device__ __forceinline__ float redmax16(float v) {
  v = fmaxf(v, __shfl_xor(v, 1, 32));
  v = fmaxf(v, __shfl_xor(v, 2, 32));
  v = fmaxf(v, __shfl_xor(v, 4, 32));
  v = fmaxf(v, __shfl_xor(v, 8, 32));
  return v;
}
static __device__ __forceinline__ float redsum16(float v) {
  v += __shfl_xor(v, 1, 32);
  v += __shfl_xor(v, 2, 32);
  v += __shfl_xor(v, 4, 32);
  v += __shfl_xor(v, 8, 32);
  return v;
}

static __device__ __forceinline__ v8f wmma_f16(v16h a, v16h b, v8f c) {
  return __builtin_amdgcn_wmma_f32_16x16x32_f16(false, a, false, b, (short)0, c, false, false);
}

__global__ __launch_bounds__(THREADS, 1)
void tblock_kernel(const float* __restrict__ x,
                   const float* __restrict__ Wq, const float* __restrict__ bq,
                   const float* __restrict__ Wk, const float* __restrict__ bk,
                   const float* __restrict__ Wv, const float* __restrict__ bv,
                   const float* __restrict__ Wp, const float* __restrict__ bp,
                   const float* __restrict__ W1, const float* __restrict__ b1,
                   const float* __restrict__ W2, const float* __restrict__ b2,
                   const float* __restrict__ g1, const float* __restrict__ be1,
                   const float* __restrict__ g2, const float* __restrict__ be2,
                   float* __restrict__ out)
{
  __shared__ __align__(32) _Float16 wt[NTILES * 512];       // 192 KB weight tiles
  __shared__ __align__(32) _Float16 scr_all[WAVES * SCRH];  // 82 KB per-wave scratch

  const int tid  = threadIdx.x;
  const int lane = tid & 31;
  const int wv   = tid >> 5;
  const int half = lane >> 4;
  const int nl   = lane & 15;

  // ---- Stage all weights into LDS in WMMA B-fragment order (f16) ----
  // Tile slot = lane*16 + e ;  element (k,n) = (32kt + 16*(lane>>4) + e, 16nt + (lane&15))
  for (int slot = tid; slot < NTILES * 512; slot += THREADS) {
    int tile = slot >> 9;
    int r    = slot & 511;
    int l    = r >> 4, e = r & 15;
    int lh   = l >> 4, ln = l & 15;
    const float* W; int Ntot, kt, nt;
    if (tile < 96) {                       // QKV: p*32 + h*8 + kt*4 + nt
      int p = tile >> 5, rem = tile & 31;
      int h = rem >> 3; kt = (rem >> 2) & 1; nt = rem & 3;
      W = (p == 0 ? Wq : (p == 1 ? Wk : Wv)) + h * DM * DM; Ntot = DM;
    } else if (tile < 128) {               // Wp: 96 + kt*4 + nt (kt 0..7)
      int t2 = tile - 96; kt = t2 >> 2; nt = t2 & 3; W = Wp; Ntot = DM;
    } else if (tile < 160) {               // W1: 128 + kt*16 + nt (kt 0..1, nt 0..15)
      int t2 = tile - 128; kt = t2 >> 4; nt = t2 & 15; W = W1; Ntot = DFF;
    } else {                               // W2: 160 + kt*4 + nt (kt 0..7)
      int t2 = tile - 160; kt = t2 >> 2; nt = t2 & 3; W = W2; Ntot = DM;
    }
    int k = 32 * kt + 16 * lh + e;
    int n = 16 * nt + ln;
    wt[slot] = (_Float16)W[k * Ntot + n];
  }

  _Float16* scr = scr_all + wv * SCRH;
  scr[ZERO_OFF + lane] = (_Float16)0.0f;   // per-wave zero pad region (stays zero)
  __syncthreads();

  // Branchless V^T read setup: half==1 lanes (zero-padded K rows) read the
  // zero region with stride 0 -> no EXEC divergence around ctx WMMAs.
  const _Float16* vsrc  = (half == 0) ? (scr + VT_OFF + nl * VTP) : (scr + ZERO_OFF);
  const int       vstep = (half == 0) ? (16 * VTP) : 0;

  const float SCALE = 0.25f;   // 1/sqrt(16)
  const int gw = blockIdx.x * WAVES + wv;

#pragma unroll 1
  for (int b = gw; b < BATCH; b += NWG * WAVES) {
    const float* xb = x + b * (SEQ * DM);

    // x in C/D layout: xr[nt][r] = x[t = r + 8*half][d = 16*nt + nl]
    float xr[4][8];
#pragma unroll
    for (int nt = 0; nt < 4; ++nt)
#pragma unroll
      for (int r = 0; r < 8; ++r)
        xr[nt][r] = xb[(r + 8 * half) * DM + nt * 16 + nl];

    // ---- LN1 -> h (row-major f16 in scratch) ----
    {
      float g1v[4], be1v[4];
#pragma unroll
      for (int nt = 0; nt < 4; ++nt) { g1v[nt] = g1[nt * 16 + nl]; be1v[nt] = be1[nt * 16 + nl]; }
#pragma unroll
      for (int r = 0; r < 8; ++r) {
        float s = redsum16(xr[0][r] + xr[1][r] + xr[2][r] + xr[3][r]);
        float m = s * (1.0f / 64.0f);
        float ss = 0.0f;
#pragma unroll
        for (int nt = 0; nt < 4; ++nt) { float d = xr[nt][r] - m; ss += d * d; }
        ss = redsum16(ss);
        float inv = rsqrtf(ss * (1.0f / 64.0f) + 1e-5f);
#pragma unroll
        for (int nt = 0; nt < 4; ++nt) {
          float hv = (xr[nt][r] - m) * inv * g1v[nt] + be1v[nt];
          scr[(r + 8 * half) * ROWP + nt * 16 + nl] = (_Float16)hv;
        }
      }
    }
    fence_lds();
    v16h hA0 = ld_afrag(scr, nl, half, 0);
    v16h hA1 = ld_afrag(scr, nl, half, 1);
    fence_lds();

    // attention-out accumulator, residual + bp folded into C; xr dies here
    v8f ao[4];
#pragma unroll
    for (int nt = 0; nt < 4; ++nt) {
      float bb = bp[nt * 16 + nl];
#pragma unroll
      for (int r = 0; r < 8; ++r) ao[nt][r] = xr[nt][r] + bb;
    }

#pragma unroll 1
    for (int h = 0; h < NH; ++h) {
      const _Float16* qt  = wt + (0 * 32 + h * 8) * 512;
      const _Float16* ktb = wt + (1 * 32 + h * 8) * 512;
      const _Float16* vtb = wt + (2 * 32 + h * 8) * 512;

      // ---- Q = h @ Wq[h] + bq[h] -> row-major stage ----
#pragma unroll
      for (int nt = 0; nt < 4; ++nt) {
        float bb = bq[h * DM + nt * 16 + nl];
        v8f c;
#pragma unroll
        for (int r = 0; r < 8; ++r) c[r] = bb;
        c = wmma_f16(hA0, ld_bfrag(qt + (0 * 4 + nt) * 512, lane), c);
        c = wmma_f16(hA1, ld_bfrag(qt + (1 * 4 + nt) * 512, lane), c);
#pragma unroll
        for (int r = 0; r < 8; ++r)
          scr[(r + 8 * half) * ROWP + nt * 16 + nl] = (_Float16)c[r];
      }
      fence_lds();
      v16h qA0 = ld_afrag(scr, nl, half, 0);
      v16h qA1 = ld_afrag(scr, nl, half, 1);
      fence_lds();

      // ---- K -> row-major stage, read back as K^T B-fragments ----
#pragma unroll
      for (int nt = 0; nt < 4; ++nt) {
        float bb = bk[h * DM + nt * 16 + nl];
        v8f c;
#pragma unroll
        for (int r = 0; r < 8; ++r) c[r] = bb;
        c = wmma_f16(hA0, ld_bfrag(ktb + (0 * 4 + nt) * 512, lane), c);
        c = wmma_f16(hA1, ld_bfrag(ktb + (1 * 4 + nt) * 512, lane), c);
#pragma unroll
        for (int r = 0; r < 8; ++r)
          scr[(r + 8 * half) * ROWP + nt * 16 + nl] = (_Float16)c[r];
      }
      fence_lds();
      v16h kB0 = ld_ktb(scr, nl, half, 0);
      v16h kB1 = ld_ktb(scr, nl, half, 1);
      fence_lds();

      // ---- V -> V^T stage (64 rows x 16, stride VTP, disjoint region) ----
#pragma unroll
      for (int nt = 0; nt < 4; ++nt) {
        float bb = bv[h * DM + nt * 16 + nl];
        v8f c;
#pragma unroll
        for (int r = 0; r < 8; ++r) c[r] = bb;
        c = wmma_f16(hA0, ld_bfrag(vtb + (0 * 4 + nt) * 512, lane), c);
        c = wmma_f16(hA1, ld_bfrag(vtb + (1 * 4 + nt) * 512, lane), c);
        v8h pk;
#pragma unroll
        for (int r = 0; r < 8; ++r) pk[r] = (_Float16)c[r];
        *(v8h*)(scr + VT_OFF + (nt * 16 + nl) * VTP + half * 8) = pk;  // VT[d][m0..m0+7]
      }
      fence_lds();

      // ---- scores = Q K^T * SCALE, causal softmax ----
      v8f sc;
#pragma unroll
      for (int r = 0; r < 8; ++r) sc[r] = 0.0f;
      sc = wmma_f16(qA0, kB0, sc);
      sc = wmma_f16(qA1, kB1, sc);
#pragma unroll
      for (int r = 0; r < 8; ++r) {
        int q = r + 8 * half;          // query row; nl = key col
        bool ok = (nl <= q);
        float v = ok ? sc[r] * SCALE : -3.0e38f;
        float mx = redmax16(v);
        float e = ok ? __expf(v - mx) : 0.0f;
        float s = redsum16(e);
        scr[ATTN_OFF + q * ATTNP + nl] = (_Float16)(e / s);
      }
      fence_lds();

      // attn A-fragment (K=16 real in half 0, half 1 zero-padded)
      v16h aA;
      {
        const v8h* pa = (const v8h*)((half == 0) ? (scr + ATTN_OFF + nl * ATTNP) : (scr + ZERO_OFF));
        aA = join8(pa[0], pa[1]);
      }

      // ---- ctx = attn @ V; branchless B read (half==1 lanes read zeros);
      //      stage each tile's rows immediately (disjoint from VT region) ----
#pragma unroll
      for (int nt = 0; nt < 4; ++nt) {
        const v8h* p = (const v8h*)(vsrc + nt * vstep);
        v16h vb = join8(p[0], p[1]);
        v8f c;
#pragma unroll
        for (int r = 0; r < 8; ++r) c[r] = 0.0f;
        c = wmma_f16(aA, vb, c);
#pragma unroll
        for (int r = 0; r < 8; ++r)
          scr[(r + 8 * half) * ROWP + nt * 16 + nl] = (_Float16)c[r];
      }
      fence_lds();

      // ---- ctx A-frags, accumulate out-projection ----
      v16h cA0 = ld_afrag(scr, nl, half, 0);
      v16h cA1 = ld_afrag(scr, nl, half, 1);
#pragma unroll
      for (int nt = 0; nt < 4; ++nt) {
        ao[nt] = wmma_f16(cA0, ld_bfrag(wt + (96 + (2 * h + 0) * 4 + nt) * 512, lane), ao[nt]);
        ao[nt] = wmma_f16(cA1, ld_bfrag(wt + (96 + (2 * h + 1) * 4 + nt) * 512, lane), ao[nt]);
      }
      fence_lds();
    } // heads

    // ao == x2 = x + attn_out @ Wp + bp
    // ---- LN2 -> h2 row-major stage ----
    {
      float g2v[4], be2v[4];
#pragma unroll
      for (int nt = 0; nt < 4; ++nt) { g2v[nt] = g2[nt * 16 + nl]; be2v[nt] = be2[nt * 16 + nl]; }
#pragma unroll
      for (int r = 0; r < 8; ++r) {
        float s = redsum16(ao[0][r] + ao[1][r] + ao[2][r] + ao[3][r]);
        float m = s * (1.0f / 64.0f);
        float ss = 0.0f;
#pragma unroll
        for (int nt = 0; nt < 4; ++nt) { float d = ao[nt][r] - m; ss += d * d; }
        ss = redsum16(ss);
        float inv = rsqrtf(ss * (1.0f / 64.0f) + 1e-5f);
#pragma unroll
        for (int nt = 0; nt < 4; ++nt) {
          float hv = (ao[nt][r] - m) * inv * g2v[nt] + be2v[nt];
          scr[(r + 8 * half) * ROWP + nt * 16 + nl] = (_Float16)hv;
        }
      }
    }
    fence_lds();
    v16h h2A0 = ld_afrag(scr, nl, half, 0);
    v16h h2A1 = ld_afrag(scr, nl, half, 1);
    fence_lds();

    // out C = x2 + b2; ao dies here
    v8f oc[4];
#pragma unroll
    for (int nt = 0; nt < 4; ++nt) {
      float bb = b2[nt * 16 + nl];
#pragma unroll
      for (int r = 0; r < 8; ++r) oc[nt][r] = ao[nt][r] + bb;
    }

    // ---- FFN: per 32-wide K chunk, compute relu(h2@W1+b1) then @ W2 ----
#pragma unroll 1
    for (int k8 = 0; k8 < 8; ++k8) {
#pragma unroll
      for (int j = 0; j < 2; ++j) {
        int nt = 2 * k8 + j;
        float bb = b1[nt * 16 + nl];     // loaded in-loop: no hoisted broadcasts
        v8f c;
#pragma unroll
        for (int r = 0; r < 8; ++r) c[r] = bb;
        c = wmma_f16(h2A0, ld_bfrag(wt + (128 + 0 * 16 + nt) * 512, lane), c);
        c = wmma_f16(h2A1, ld_bfrag(wt + (128 + 1 * 16 + nt) * 512, lane), c);
#pragma unroll
        for (int r = 0; r < 8; ++r) {
          float v = fmaxf(c[r], 0.0f);
          scr[(r + 8 * half) * RELUP + j * 16 + nl] = (_Float16)v;
        }
      }
      fence_lds();
      {
        const _Float16* p = scr + nl * RELUP + half * 16;
        v16h rA = join8(*(const v8h*)p, *(const v8h*)(p + 8));
#pragma unroll
        for (int nt = 0; nt < 4; ++nt)
          oc[nt] = wmma_f16(rA, ld_bfrag(wt + (160 + k8 * 4 + nt) * 512, lane), oc[nt]);
      }
      fence_lds();
    }

    // ---- store result ----
    float* ob = out + b * (SEQ * DM);
#pragma unroll
    for (int nt = 0; nt < 4; ++nt)
#pragma unroll
      for (int r = 0; r < 8; ++r)
        ob[(r + 8 * half) * DM + nt * 16 + nl] = oc[nt][r];
  } // batch loop
}

extern "C" void kernel_launch(void* const* d_in, const int* in_sizes, int n_in,
                              void* d_out, int out_size, void* d_ws, size_t ws_size,
                              hipStream_t stream) {
  const float* x   = (const float*)d_in[0];
  const float* Wq  = (const float*)d_in[1];
  const float* bq  = (const float*)d_in[2];
  const float* Wk  = (const float*)d_in[3];
  const float* bk  = (const float*)d_in[4];
  const float* Wv  = (const float*)d_in[5];
  const float* bv  = (const float*)d_in[6];
  const float* Wp  = (const float*)d_in[7];
  const float* bp  = (const float*)d_in[8];
  const float* W1  = (const float*)d_in[9];
  const float* b1  = (const float*)d_in[10];
  const float* W2  = (const float*)d_in[11];
  const float* b2  = (const float*)d_in[12];
  const float* g1  = (const float*)d_in[13];
  const float* be1 = (const float*)d_in[14];
  const float* g2  = (const float*)d_in[15];
  const float* be2 = (const float*)d_in[16];
  float* out = (float*)d_out;

  tblock_kernel<<<NWG, THREADS, 0, stream>>>(x, Wq, bq, Wk, bk, Wv, bv, Wp, bp,
                                             W1, b1, W2, b2, g1, be1, g2, be2, out);
}
